// GATLayer_31688268710357
// MI455X (gfx1250) — matrix operand order repaired and checked by reading, not compiled, not run on hardware
//
#include <hip/hip_runtime.h>
#include <stdint.h>

#define GN 8192
#define GD 256

typedef __attribute__((ext_vector_type(16))) __bf16   v16bf;
typedef __attribute__((ext_vector_type(8)))  float    v8f;
typedef __attribute__((ext_vector_type(4)))  int      v4i;
typedef __attribute__((ext_vector_type(4)))  float    v4f;
typedef __attribute__((ext_vector_type(4)))  unsigned v4u;

// ---------- bf16 helpers (round-to-nearest-even) ----------
static __device__ __forceinline__ unsigned short f2bf(float f) {
  unsigned u = __builtin_bit_cast(unsigned, f);
  u += 0x7FFFu + ((u >> 16) & 1u);
  return (unsigned short)(u >> 16);
}
static __device__ __forceinline__ float bf2f(unsigned short h) {
  unsigned u = ((unsigned)h) << 16;
  return __builtin_bit_cast(float, u);
}
static __device__ __forceinline__ unsigned packbf2(float a, float b) {
  return (unsigned)f2bf(a) | ((unsigned)f2bf(b) << 16);
}

union TileU { v4u q[2]; unsigned u[8]; v16bf v; };

// A-matrix 16x32 bf16 tile from row-major (K-contiguous) source.
// lane: M = lane&15 ; K = {g*8..g*8+7} U {16+g*8..16+g*8+7}, g = lane>>4
static __device__ __forceinline__ v16bf load_a16(const unsigned short* base, int stride, int lane) {
  int r = lane & 15, g = lane >> 4;
  const unsigned short* p = base + (size_t)r * stride + g * 8;
  TileU t;
  t.q[0] = *(const v4u*)p;
  t.q[1] = *(const v4u*)(p + 16);
  return t.v;
}
// B-matrix 32x16 bf16 tile; source is B^T row-major (row = n, K contiguous).
// lane: N = lane&15 ; VGPR0-3 = K g*8..+7 ; VGPR4-7 = K 16+g*8..+7
static __device__ __forceinline__ v16bf load_b16(const unsigned short* base, int stride, int lane) {
  int r = lane & 15, g = lane >> 4;
  const unsigned short* p = base + (size_t)r * stride + g * 8;
  TileU t;
  t.q[0] = *(const v4u*)p;
  t.q[1] = *(const v4u*)(p + 16);
  return t.v;
}

static __device__ __forceinline__ v8f wmma_bf16(v16bf a, v16bf b, v8f c) {
  return __builtin_amdgcn_wmma_f32_16x16x32_bf16(false, a, false, b, (short)0, c, false, false);
}

// ---------- kernel 1: fp32 -> bf16 convert ----------
__global__ void cvt_kernel(const float* __restrict__ src, unsigned short* __restrict__ dst, int n) {
  int i = blockIdx.x * 256 + threadIdx.x;
  if (i < n) dst[i] = f2bf(src[i]);
}

// ---------- kernel 2: h = X @ W^T + b, stored transposed in bf16: hT[d][m] ----------
// 512 blocks x 128 threads (4 waves). Each wave: 16 rows x 64 cols, K-loop over 256.
__global__ __launch_bounds__(128) void gemm_h_kernel(const unsigned short* __restrict__ Xb,
                                                     const unsigned short* __restrict__ Wb,
                                                     const float* __restrict__ bias,
                                                     unsigned short* __restrict__ hT) {
  int lane = threadIdx.x & 31, warp = threadIdx.x >> 5;
  int gid = blockIdx.x * 4 + warp;        // 0..2047 waves
  int mbase = (gid >> 2) * 16;            // row tile
  int n64 = (gid & 3) * 64;               // 64-col strip
  int g = lane >> 4, r15 = lane & 15;

  v8f acc[4];
#pragma unroll
  for (int t = 0; t < 4; ++t)
#pragma unroll
    for (int i = 0; i < 8; ++i) acc[t][i] = 0.f;

#pragma unroll
  for (int k = 0; k < GD; k += 32) {
    v16bf a = load_a16(Xb + (size_t)mbase * GD + k, GD, lane);
#pragma unroll
    for (int t = 0; t < 4; ++t) {
      // B[k][d] = W[d][k]  ->  B^T row-major source is W itself
      v16bf b = load_b16(Wb + (size_t)(n64 + t * 16) * GD + k, GD, lane);
      acc[t] = wmma_bf16(a, b, acc[t]);
    }
  }

#pragma unroll
  for (int t = 0; t < 4; ++t) {
    int dcol = n64 + t * 16 + r15;
    float bv = bias[dcol];
    union { unsigned short s[8]; v4u q; } pk;
#pragma unroll
    for (int r = 0; r < 8; ++r) pk.s[r] = f2bf(acc[t][r] + bv);   // rows m = 8g..8g+7
    *(v4u*)(hT + (size_t)dcol * GN + mbase + 8 * g) = pk.q;
  }
}

// ---------- kernel 3: e1/e2 = h @ we + be (reads hT columns, coalesced) ----------
__global__ __launch_bounds__(256) void e_kernel(const unsigned short* __restrict__ hT,
                                                const float* __restrict__ we1,
                                                const float* __restrict__ we2,
                                                const float* __restrict__ be1,
                                                const float* __restrict__ be2,
                                                float* __restrict__ e1,
                                                float* __restrict__ e2) {
  int n = blockIdx.x * 256 + threadIdx.x;
  float a1 = 0.f, a2 = 0.f;
#pragma unroll 4
  for (int d = 0; d < GD; ++d) {
    float hv = bf2f(hT[(size_t)d * GN + n]);
    a1 += hv * we1[d];
    a2 += hv * we2[d];
  }
  e1[n] = a1 + be1[0];
  e2[n] = a2 + be2[0];
}

// ---------- kernel 4: fused masked-softmax attention aggregate ----------
// out[i,:] = (sum_j p_ij * h[j,:]) / (sum_j p_ij),  p_ij = adj ? exp(leaky(e1_i+e2_j)) : 0
#define LDS_PITCH 40            // ushorts per hT row slice (64B data + 16B pad)
#define BUF_US    (256 * LDS_PITCH)
#define BUF_BYTES (BUF_US * 2)

static __device__ __forceinline__ void stage_async(const unsigned short* __restrict__ hT,
                                                   int jbase, unsigned ldsbyte, int tid) {
  // copy hT[d][jbase..jbase+31] for d=0..255 into LDS rows of LDS_PITCH ushorts
#pragma unroll
  for (int i = 0; i < 8; ++i) {
    int c = tid + i * 128;              // 1024 chunks of 16B
    int d = c >> 2, part = c & 3;
    unsigned long long ga =
        (unsigned long long)(uintptr_t)(hT + (size_t)d * GN + jbase + part * 8);
    unsigned lo = ldsbyte + (unsigned)(d * (LDS_PITCH * 2) + part * 16);
    asm volatile("global_load_async_to_lds_b128 %0, %1, off" :: "v"(lo), "v"(ga) : "memory");
  }
}

static __device__ __forceinline__ float pval(float e1r, float e2v, int edge, float& lsum) {
  float s = e1r + e2v;
  s = s >= 0.f ? s : 0.01f * s;         // leaky relu
  float p = edge > 0 ? __expf(s) : 0.f; // mask -> exact zero (replaces exp(NEG_INF))
  lsum += p;
  return p;
}

__global__ __launch_bounds__(128) void gat_flash_kernel(const unsigned short* __restrict__ hT,
                                                        const float* __restrict__ e1,
                                                        const float* __restrict__ e2,
                                                        const int* __restrict__ adj,
                                                        float* __restrict__ out) {
  __shared__ unsigned short smem[2 * BUF_US];  // 40 KB, double-buffered B stage

  int tid = threadIdx.x;
  int lane = tid & 31, warp = tid >> 5;
  int g = lane >> 4, r15 = lane & 15;
  int rowb = blockIdx.x * 64 + warp * 16;      // this wave's 16-row tile
  int row = rowb + r15;                        // P-row owned by this lane (dup at lane+16)

  float e1r = e1[row];
  unsigned lds0 = (unsigned)(uintptr_t)(void*)smem;

  v8f acc[16];
#pragma unroll
  for (int t = 0; t < 16; ++t)
#pragma unroll
    for (int i = 0; i < 8; ++i) acc[t][i] = 0.f;
  float lsum = 0.f;

  stage_async(hT, 0, lds0, tid);
  asm volatile("s_wait_asynccnt 0x0" ::: "memory");
  __syncthreads();

  const int NSTEP = GN / 32;
  for (int jb = 0; jb < NSTEP; ++jb) {
    int jbase = jb * 32;
    if (jb + 1 < NSTEP)
      stage_async(hT, jbase + 32, lds0 + (unsigned)(((jb + 1) & 1) * BUF_BYTES), tid);

    // ---- build P A-tile (16x32 bf16) in registers ----
    const int* arow = adj + (size_t)row * GN + jbase;
    v4i m0 = __builtin_nontemporal_load((const v4i*)(arow + g * 8));
    v4i m1 = __builtin_nontemporal_load((const v4i*)(arow + g * 8 + 4));
    v4i m2 = __builtin_nontemporal_load((const v4i*)(arow + 16 + g * 8));
    v4i m3 = __builtin_nontemporal_load((const v4i*)(arow + 20 + g * 8));
    const float* ep = e2 + jbase;
    v4f f0 = *(const v4f*)(ep + g * 8);
    v4f f1 = *(const v4f*)(ep + g * 8 + 4);
    v4f f2 = *(const v4f*)(ep + 16 + g * 8);
    v4f f3 = *(const v4f*)(ep + 20 + g * 8);

    TileU au;
    au.u[0] = packbf2(pval(e1r, f0.x, m0.x, lsum), pval(e1r, f0.y, m0.y, lsum));
    au.u[1] = packbf2(pval(e1r, f0.z, m0.z, lsum), pval(e1r, f0.w, m0.w, lsum));
    au.u[2] = packbf2(pval(e1r, f1.x, m1.x, lsum), pval(e1r, f1.y, m1.y, lsum));
    au.u[3] = packbf2(pval(e1r, f1.z, m1.z, lsum), pval(e1r, f1.w, m1.w, lsum));
    au.u[4] = packbf2(pval(e1r, f2.x, m2.x, lsum), pval(e1r, f2.y, m2.y, lsum));
    au.u[5] = packbf2(pval(e1r, f2.z, m2.z, lsum), pval(e1r, f2.w, m2.w, lsum));
    au.u[6] = packbf2(pval(e1r, f3.x, m3.x, lsum), pval(e1r, f3.y, m3.y, lsum));
    au.u[7] = packbf2(pval(e1r, f3.z, m3.z, lsum), pval(e1r, f3.w, m3.w, lsum));
    v16bf a = au.v;

    // ---- O += P @ H_block : 16 d-tiles from LDS ----
    const unsigned short* buf = smem + (jb & 1) * BUF_US;
#pragma unroll
    for (int t = 0; t < 16; ++t) {
      v16bf b = load_b16(buf + t * 16 * LDS_PITCH, LDS_PITCH, lane);
      acc[t] = wmma_bf16(a, b, acc[t]);
    }

    asm volatile("s_wait_asynccnt 0x0" ::: "memory");
    __syncthreads();
  }

  // ---- normalize and store ----
  float ltot = lsum + __shfl_xor(lsum, 16, 32);   // full row sum (lane pair L, L+16)
#pragma unroll
  for (int r = 0; r < 8; ++r) {
    float lr = __shfl(ltot, g * 8 + r, 32);       // l for accumulator row M = r + 8g
    float inv = 1.0f / lr;
    float* orow = out + (size_t)(rowb + 8 * g + r) * GD + r15;
#pragma unroll
    for (int t = 0; t < 16; ++t) orow[t * 16] = acc[t][r] * inv;
  }
}

// ---------- launch ----------
extern "C" void kernel_launch(void* const* d_in, const int* in_sizes, int n_in,
                              void* d_out, int out_size, void* d_ws, size_t ws_size,
                              hipStream_t stream) {
  const float* X   = (const float*)d_in[0];
  const int*   adj = (const int*)d_in[1];
  const float* W   = (const float*)d_in[2];
  const float* b   = (const float*)d_in[3];
  const float* we1 = (const float*)d_in[4];
  const float* be1 = (const float*)d_in[5];
  const float* we2 = (const float*)d_in[6];
  const float* be2 = (const float*)d_in[7];
  float* out = (float*)d_out;

  char* ws = (char*)d_ws;
  unsigned short* Xb = (unsigned short*)(ws);                         // 4 MiB
  unsigned short* Wb = (unsigned short*)(ws + (4u << 20));            // 128 KiB (pad 256K)
  unsigned short* hT = (unsigned short*)(ws + (4u << 20) + (256u << 10)); // 4 MiB
  float* e1 = (float*)(ws + (8u << 20) + (512u << 10));               // 32 KiB
  float* e2 = e1 + GN;                                                // 32 KiB

  cvt_kernel<<<(GN * GD + 255) / 256, 256, 0, stream>>>(X, Xb, GN * GD);
  cvt_kernel<<<(GD * GD + 255) / 256, 256, 0, stream>>>(W, Wb, GD * GD);
  gemm_h_kernel<<<(GN / 16) * (GD / 64) / 4, 128, 0, stream>>>(Xb, Wb, b, hT);
  e_kernel<<<GN / 256, 256, 0, stream>>>(hT, we1, we2, be1, be2, e1, e2);
  gat_flash_kernel<<<GN / 64, 128, 0, stream>>>(hT, e1, e2, adj, out);
}
